// BitWhiskerLM_60404420051180
// MI455X (gfx1250) — matrix operand order repaired and checked by reading, not compiled, not run on hardware
//
#include <hip/hip_runtime.h>

// ---------------- problem constants ----------------
constexpr int V  = 32000;
constexpr int B  = 2;
constexpr int S  = 1024;
constexpr int D  = 1024;
constexpr int H  = 16;
constexpr int HD = 64;
constexpr int L  = 4;
constexpr int FF = 2816;
constexpr int M  = B * S;          // 2048 token rows
constexpr float EPS = 1e-6f;

typedef __attribute__((ext_vector_type(8)))  unsigned short ushort8;
typedef __attribute__((ext_vector_type(16))) unsigned short v16u;
typedef __attribute__((ext_vector_type(16))) __bf16        v16bf;
typedef __attribute__((ext_vector_type(8)))  float         v8f;

__device__ __forceinline__ unsigned short f2bf(float f) {
  unsigned int u = __float_as_uint(f);
  u += 0x7FFFu + ((u >> 16) & 1u);   // round-to-nearest-even
  return (unsigned short)(u >> 16);
}

// Combine two 8-half chunks into one 16-half WMMA fragment register block.
__device__ __forceinline__ v16bf make_frag(ushort8 lo, ushort8 hi) {
  union { v16u u; v16bf b; } cv;
#pragma unroll
  for (int j = 0; j < 8; ++j) { cv.u[j] = lo[j]; cv.u[j + 8] = hi[j]; }
  return cv.b;
}

// gfx1250 async memory->LDS DMA (ASYNCcnt-tracked), per ISA 15.18.3.
__device__ __forceinline__ void async_b128(unsigned lds, unsigned long long ga) {
  asm volatile("global_load_async_to_lds_b128 %0, %1, off"
               :: "v"(lds), "v"(ga) : "memory");
}
__device__ __forceinline__ void wait_async0() {
  asm volatile("s_wait_asynccnt 0x0" ::: "memory");
}

// ---------------- elementwise / prep kernels ----------------
__global__ __launch_bounds__(256) void cast_f32_bf16_k(const float* __restrict__ src,
                                                       unsigned short* __restrict__ dst,
                                                       size_t n) {
  size_t i = (size_t)blockIdx.x * blockDim.x + threadIdx.x;
  size_t stride = (size_t)gridDim.x * blockDim.x;
  for (; i < n; i += stride) dst[i] = f2bf(src[i]);
}

// src: K x N f32 row-major  ->  dst: N x K bf16 row-major (tiled transpose)
__global__ __launch_bounds__(256) void transcast_k(const float* __restrict__ src,
                                                   unsigned short* __restrict__ dst,
                                                   int K, int N) {
  __shared__ float tile[32][33];
  const int x  = threadIdx.x & 31;
  const int y0 = threadIdx.x >> 5;      // 0..7
#pragma unroll
  for (int j = 0; j < 4; ++j) {
    int k = blockIdx.y * 32 + y0 + j * 8;
    tile[y0 + j * 8][x] = src[(size_t)k * N + blockIdx.x * 32 + x];
  }
  __syncthreads();
#pragma unroll
  for (int j = 0; j < 4; ++j) {
    int n = blockIdx.x * 32 + y0 + j * 8;
    dst[(size_t)n * K + blockIdx.y * 32 + x] = f2bf(tile[x][y0 + j * 8]);
  }
}

__global__ __launch_bounds__(256) void embed_k(const int* __restrict__ tok,
                                               const float* __restrict__ emb,
                                               float* __restrict__ h) {
  int m = blockIdx.x;
  int t = tok[m];
  const float4* srcv = (const float4*)(emb + (size_t)t * D);
  float4* dstv = (float4*)(h + (size_t)m * D);
  dstv[threadIdx.x] = srcv[threadIdx.x];
}

__global__ __launch_bounds__(256) void rmsnorm_k(const float* __restrict__ x,
                                                 const float* __restrict__ w,
                                                 unsigned short* __restrict__ out) {
  __shared__ float red[256];
  const int m = blockIdx.x;
  const int t = threadIdx.x;
  const float* xr = x + (size_t)m * D;
  float v0 = xr[t], v1 = xr[t + 256], v2 = xr[t + 512], v3 = xr[t + 768];
  red[t] = v0 * v0 + v1 * v1 + v2 * v2 + v3 * v3;
  __syncthreads();
  for (int s2 = 128; s2 > 0; s2 >>= 1) {
    if (t < s2) red[t] += red[t + s2];
    __syncthreads();
  }
  float rs = rsqrtf(red[0] / (float)D + EPS);
  unsigned short* o = out + (size_t)m * D;
  o[t]       = f2bf(v0 * rs * w[t]);
  o[t + 256] = f2bf(v1 * rs * w[t + 256]);
  o[t + 512] = f2bf(v2 * rs * w[t + 512]);
  o[t + 768] = f2bf(v3 * rs * w[t + 768]);
}

// RoPE: reads qkvf (3 x M x D f32), writes
//   qbT,kbT : [b*H+h][s][hd] bf16   (GEMM A/B rows, lda=HD, linear bh stride)
//   vbT     : [b*H+h][hd][s] bf16   (pre-transposed B for the PV GEMM)
__global__ __launch_bounds__(256) void rope_qkv_k(const float* __restrict__ qkvf,
                                                  unsigned short* __restrict__ qbT,
                                                  unsigned short* __restrict__ kbT,
                                                  unsigned short* __restrict__ vbT) {
  const int m = blockIdx.x;
  const int b = m >> 10;             // m / S
  const int s = m & (S - 1);
  const float* qf = qkvf;
  const float* kf = qkvf + (size_t)M * D;
  const float* vf = qkvf + (size_t)2 * M * D;
  for (int p = threadIdx.x; p < D / 2; p += 256) {
    int h = p >> 5;                  // head
    int d = p & 31;                  // pair index (HD/2 = 32)
    float inv = __powf(10000.0f, -(float)(2 * d) * (1.0f / 64.0f));
    float ang = (float)s * inv;
    float cs = __cosf(ang), sn = __sinf(ang);
    size_t src = (size_t)m * D + 2 * p;
    size_t bh  = (size_t)b * H + h;
    size_t qkDst = (bh * S + s) * HD + 2 * d;
    float q0 = qf[src], q1 = qf[src + 1];
    qbT[qkDst]     = f2bf(q0 * cs - q1 * sn);
    qbT[qkDst + 1] = f2bf(q0 * sn + q1 * cs);
    float k0 = kf[src], k1 = kf[src + 1];
    kbT[qkDst]     = f2bf(k0 * cs - k1 * sn);
    kbT[qkDst + 1] = f2bf(k0 * sn + k1 * cs);
    size_t vDst = (bh * HD + 2 * d) * S + s;
    vbT[vDst]     = f2bf(vf[src]);
    vbT[vDst + S] = f2bf(vf[src + 1]);
  }
}

// Causal softmax over one score row; repacks probs as bf16 in place.
__global__ __launch_bounds__(256) void softmax_causal_k(float* __restrict__ sc) {
  __shared__ float red[256];
  const int row = blockIdx.x & (S - 1);
  float* r = sc + (size_t)blockIdx.x * S;
  const int t = threadIdx.x;
  float v[4];
  float mx = -1e30f;
#pragma unroll
  for (int q = 0; q < 4; ++q) {
    int j = t + q * 256;
    float x = (j <= row) ? r[j] : -1e30f;
    v[q] = x;
    mx = fmaxf(mx, x);
  }
  red[t] = mx; __syncthreads();
  for (int s2 = 128; s2 > 0; s2 >>= 1) {
    if (t < s2) red[t] = fmaxf(red[t], red[t + s2]);
    __syncthreads();
  }
  mx = red[0]; __syncthreads();
  float sum = 0.f;
#pragma unroll
  for (int q = 0; q < 4; ++q) {
    int j = t + q * 256;
    float e = (j <= row) ? __expf(v[q] - mx) : 0.f;
    v[q] = e; sum += e;
  }
  red[t] = sum; __syncthreads();
  for (int s2 = 128; s2 > 0; s2 >>= 1) {
    if (t < s2) red[t] += red[t + s2];
    __syncthreads();
  }
  float inv = 1.0f / red[0];
  unsigned short* p = (unsigned short*)r;   // all float reads happened pre-barrier
#pragma unroll
  for (int q = 0; q < 4; ++q) {
    int j = t + q * 256;
    p[j] = f2bf(v[q] * inv);
  }
}

// attnT [b*H+h][s][hd] f32 -> aob [b][s][h*HD+hd] bf16
__global__ __launch_bounds__(256) void attn_permcast_k(const float* __restrict__ attnT,
                                                       unsigned short* __restrict__ aob) {
  const int m = blockIdx.x;
  const int b = m >> 10;
  const int s = m & (S - 1);
  for (int d = threadIdx.x; d < D; d += 256) {
    int h = d >> 6, hd = d & 63;
    aob[(size_t)m * D + d] =
        f2bf(attnT[(((size_t)b * H + h) * S + s) * HD + hd]);
  }
}

__global__ __launch_bounds__(256) void silu_mul_k(const float* __restrict__ g1,
                                                  const float* __restrict__ g3,
                                                  unsigned short* __restrict__ gs,
                                                  size_t n) {
  size_t i = (size_t)blockIdx.x * blockDim.x + threadIdx.x;
  size_t stride = (size_t)gridDim.x * blockDim.x;
  for (; i < n; i += stride) {
    float a = g1[i];
    float s = a / (1.0f + __expf(-a));
    gs[i] = f2bf(s * g3[i]);
  }
}

__global__ __launch_bounds__(256) void add_res_k(float* __restrict__ h,
                                                 const float* __restrict__ t,
                                                 size_t n) {
  size_t i = (size_t)blockIdx.x * blockDim.x + threadIdx.x;
  size_t stride = (size_t)gridDim.x * blockDim.x;
  for (; i < n; i += stride) h[i] += t[i];
}

// ---------------- WMMA bf16 GEMM (all operands "BT": B is NxK row-major) ----
// C(MxN,f32) = alpha * A(MxK,bf16,lda) * B(NxK,bf16,ldb)^T, batched on z.
// Block tile 128x64, K-step 64, double-buffered LDS staged with CDNA5 async
// memory->LDS DMA (global_load_async_to_lds_b128, ASYNCcnt), 8 wave32 waves
// (4x2), each wave computes a 32x32 patch (8 WMMAs per K-step).
constexpr int BMT = 128, BNT = 64, BKT = 64;
constexpr int LDAS = 72;   // padded LDS row strides (halves): 144B, 16B-aligned
constexpr int LDBS = 72;

__global__ __launch_bounds__(256)
void gemm_bt_k(const unsigned short* __restrict__ A,
               const unsigned short* __restrict__ Bm,
               float* __restrict__ C,
               int Kdim, int lda, int ldb, int ldc,
               long long bsA, long long bsB, long long bsC, float alpha) {
  __shared__ alignas(16) unsigned short sA[2][BMT * LDAS];  // [row][k]
  __shared__ alignas(16) unsigned short sB[2][BNT * LDBS];  // [n][k]

  A  += (size_t)blockIdx.z * bsA;
  Bm += (size_t)blockIdx.z * bsB;
  C  += (size_t)blockIdx.z * bsC;

  const int tid  = threadIdx.x;
  const int lane = tid & 31;
  const int wv   = tid >> 5;
  const int wm   = wv >> 1;                 // 0..3 (M)
  const int wn   = wv & 1;                  // 0..1 (N)
  const int blockRow = blockIdx.y * BMT;
  const int blockCol = blockIdx.x * BNT;

  const int aRow = tid >> 1;                // 0..127
  const int aCol = (tid & 1) << 5;          // 0 or 32 halves
  const int bRow = tid >> 2;                // 0..63
  const int bCol = (tid & 3) << 4;          // 0,16,32,48 halves

  v8f c00 = {}, c01 = {}, c10 = {}, c11 = {};

  // per-thread global source / LDS destination addresses for staging
  const unsigned short* gA = A + (size_t)(blockRow + aRow) * lda + aCol;
  const unsigned short* gB = Bm + (size_t)(blockCol + bRow) * ldb + bCol;
  unsigned ldsA0 = (unsigned)(size_t)&sA[0][aRow * LDAS + aCol];
  unsigned ldsA1 = (unsigned)(size_t)&sA[1][aRow * LDAS + aCol];
  unsigned ldsB0 = (unsigned)(size_t)&sB[0][bRow * LDBS + bCol];
  unsigned ldsB1 = (unsigned)(size_t)&sB[1][bRow * LDBS + bCol];

  auto stage = [&](int kt, int buf) {
    unsigned la = buf ? ldsA1 : ldsA0;
    unsigned lb = buf ? ldsB1 : ldsB0;
    unsigned long long a = (unsigned long long)(size_t)(gA + kt);
    unsigned long long b = (unsigned long long)(size_t)(gB + kt);
    async_b128(la,      a);
    async_b128(la + 16, a + 16);
    async_b128(la + 32, a + 32);
    async_b128(la + 48, a + 48);
    async_b128(lb,      b);
    async_b128(lb + 16, b + 16);
  };

  stage(0, 0);

  const int r  = lane & 15;
  const int hi = lane >> 4;
  const int nT = Kdim / BKT;

  for (int t = 0; t < nT; ++t) {
    const int cur = t & 1;
    wait_async0();                   // this wave's DMA into buf[cur] complete
    __syncthreads();                 // whole block's buf[cur] visible; buf[cur^1] free
    if (t + 1 < nT) stage((t + 1) * BKT, cur ^ 1);   // overlaps the WMMAs below
#pragma unroll
    for (int ks = 0; ks < 2; ++ks) {
      const int ko = ks * 32;
      // A frag (16x32): lane<16 -> row r, K {0..7,16..23}; lane>=16 -> K {8..15,24..31}
      const unsigned short* pa0 = &sA[cur][(wm * 32 +      r) * LDAS + ko + hi * 8];
      const unsigned short* pa1 = &sA[cur][(wm * 32 + 16 + r) * LDAS + ko + hi * 8];
      // B frag (32x16): lane<16 -> col r, K {0..7,16..23}; lane>=16 -> K {8..15,24..31}
      const unsigned short* pb0 = &sB[cur][(wn * 32 +      r) * LDBS + ko + hi * 8];
      const unsigned short* pb1 = &sB[cur][(wn * 32 + 16 + r) * LDBS + ko + hi * 8];

      v16bf a0 = make_frag(*(const ushort8*)pa0, *(const ushort8*)(pa0 + 16));
      v16bf a1 = make_frag(*(const ushort8*)pa1, *(const ushort8*)(pa1 + 16));
      v16bf b0 = make_frag(*(const ushort8*)pb0, *(const ushort8*)(pb0 + 16));
      v16bf b1 = make_frag(*(const ushort8*)pb1, *(const ushort8*)(pb1 + 16));

      c00 = __builtin_amdgcn_wmma_f32_16x16x32_bf16(false, a0, false, b0, (short)0, c00, false, false);
      c01 = __builtin_amdgcn_wmma_f32_16x16x32_bf16(false, a0, false, b1, (short)0, c01, false, false);
      c10 = __builtin_amdgcn_wmma_f32_16x16x32_bf16(false, a1, false, b0, (short)0, c10, false, false);
      c11 = __builtin_amdgcn_wmma_f32_16x16x32_bf16(false, a1, false, b1, (short)0, c11, false, false);
    }
  }

  const int col    = lane & 15;
  const int rowAdd = (lane >> 4) * 8;
  const int row0 = blockRow + wm * 32;
  const int col0 = blockCol + wn * 32;
#pragma unroll
  for (int q = 0; q < 8; ++q) {
    C[(size_t)(row0 +      rowAdd + q) * ldc + col0 +      col] = alpha * c00[q];
    C[(size_t)(row0 +      rowAdd + q) * ldc + col0 + 16 + col] = alpha * c01[q];
    C[(size_t)(row0 + 16 + rowAdd + q) * ldc + col0 +      col] = alpha * c10[q];
    C[(size_t)(row0 + 16 + rowAdd + q) * ldc + col0 + 16 + col] = alpha * c11[q];
  }
}

// ---------------- host-side orchestration ----------------
extern "C" void kernel_launch(void* const* d_in, const int* in_sizes, int n_in,
                              void* d_out, int out_size, void* d_ws, size_t ws_size,
                              hipStream_t stream) {
  (void)in_sizes; (void)n_in; (void)out_size; (void)ws_size;

  const int*   tokens = (const int*)d_in[0];
  const float* emb    = (const float*)d_in[1];
  const float* wq     = (const float*)d_in[2];
  const float* wk     = (const float*)d_in[3];
  const float* wvw    = (const float*)d_in[4];
  const float* wo     = (const float*)d_in[5];
  const float* w1     = (const float*)d_in[6];
  const float* w2     = (const float*)d_in[7];
  const float* w3     = (const float*)d_in[8];
  const float* anw    = (const float*)d_in[9];
  const float* fnw    = (const float*)d_in[10];
  const float* finw   = (const float*)d_in[11];

  char* p = (char*)d_ws;
  auto take = [&](size_t bytes) -> void* {
    void* r = (void*)p;
    p += (bytes + 255) & ~(size_t)255;
    return r;
  };

  unsigned short* embB   = (unsigned short*)take((size_t)V * D * 2);
  unsigned short* wqkvT  = (unsigned short*)take((size_t)L * 3 * D * D * 2); // [l][{q,k,v}] NxK
  unsigned short* woT    = (unsigned short*)take((size_t)L * D * D * 2);
  unsigned short* w13T   = (unsigned short*)take((size_t)L * 2 * FF * D * 2);
  unsigned short* w2T    = (unsigned short*)take((size_t)L * D * FF * 2);
  float*          hbuf   = (float*)take((size_t)M * D * 4);
  unsigned short* hn     = (unsigned short*)take((size_t)M * D * 2);
  unsigned short* fn     = (unsigned short*)take((size_t)M * D * 2);
  float*          qkvf   = (float*)take((size_t)3 * M * D * 4);
  unsigned short* qbT    = (unsigned short*)take((size_t)M * D * 2);
  unsigned short* kbT    = (unsigned short*)take((size_t)M * D * 2);
  unsigned short* vbT    = (unsigned short*)take((size_t)M * D * 2);
  float*          sc     = (float*)take((size_t)B * H * S * S * 4); // hosts g1/g3 too
  float*          attnT  = (float*)take((size_t)M * D * 4);
  unsigned short* aob    = (unsigned short*)take((size_t)M * D * 2);
  float*          tmp    = (float*)take((size_t)M * D * 4);
  unsigned short* gs     = (unsigned short*)take((size_t)M * FF * 2);
  float* g1 = sc;
  float* g3 = sc + (size_t)M * FF;   // disjoint lifetime vs scores

  const dim3 blk(256);

  // --- weight prep: cast (+ transpose to NxK) ---
  cast_f32_bf16_k<<<8192, blk, 0, stream>>>(emb, embB, (size_t)V * D);
  for (int l = 0; l < L; ++l) {
    transcast_k<<<dim3(D / 32, D / 32), blk, 0, stream>>>(
        wq + (size_t)l * D * D, wqkvT + (size_t)(l * 3 + 0) * D * D, D, D);
    transcast_k<<<dim3(D / 32, D / 32), blk, 0, stream>>>(
        wk + (size_t)l * D * D, wqkvT + (size_t)(l * 3 + 1) * D * D, D, D);
    transcast_k<<<dim3(D / 32, D / 32), blk, 0, stream>>>(
        wvw + (size_t)l * D * D, wqkvT + (size_t)(l * 3 + 2) * D * D, D, D);
    transcast_k<<<dim3(D / 32, D / 32), blk, 0, stream>>>(
        wo + (size_t)l * D * D, woT + (size_t)l * D * D, D, D);
    transcast_k<<<dim3(FF / 32, D / 32), blk, 0, stream>>>(
        w1 + (size_t)l * D * FF, w13T + (size_t)(l * 2 + 0) * FF * D, D, FF);
    transcast_k<<<dim3(FF / 32, D / 32), blk, 0, stream>>>(
        w3 + (size_t)l * D * FF, w13T + (size_t)(l * 2 + 1) * FF * D, D, FF);
    transcast_k<<<dim3(D / 32, FF / 32), blk, 0, stream>>>(
        w2 + (size_t)l * FF * D, w2T + (size_t)l * D * FF, FF, D);
  }

  embed_k<<<M, blk, 0, stream>>>(tokens, emb, hbuf);

  const float scaleQK = 0.125f;   // 1/sqrt(HD)

  for (int l = 0; l < L; ++l) {
    // attn rmsnorm -> bf16
    rmsnorm_k<<<M, blk, 0, stream>>>(hbuf, anw + (size_t)l * D, hn);

    // fused QKV: z = 3 over {Wq,Wk,Wv}^T, outputs contiguous q|k|v (each MxD)
    gemm_bt_k<<<dim3(D / BNT, M / BMT, 3), blk, 0, stream>>>(
        hn, wqkvT + (size_t)l * 3 * D * D, qkvf,
        D, D, D, D, 0LL, (long long)D * D, (long long)M * D, 1.0f);

    // RoPE + bf16 repack into attention-friendly layouts
    rope_qkv_k<<<M, blk, 0, stream>>>(qkvf, qbT, kbT, vbT);

    // scores = Q K^T * scale, batched over 32 heads
    gemm_bt_k<<<dim3(S / BNT, S / BMT, B * H), blk, 0, stream>>>(
        qbT, kbT, sc, HD, HD, HD, S,
        (long long)S * HD, (long long)S * HD, (long long)S * S, scaleQK);

    // causal softmax, probs repacked bf16 in place
    softmax_causal_k<<<B * H * S, blk, 0, stream>>>(sc);

    // attnT = P V, batched over 32 heads (P: lda = 2S halves; V^T rows: HD x S)
    gemm_bt_k<<<dim3(HD / BNT, S / BMT, B * H), blk, 0, stream>>>(
        (const unsigned short*)sc, vbT, attnT, S, 2 * S, S, HD,
        (long long)2 * S * S, (long long)HD * S, (long long)S * HD, 1.0f);

    // h += attn @ Wo
    attn_permcast_k<<<M, blk, 0, stream>>>(attnT, aob);
    gemm_bt_k<<<dim3(D / BNT, M / BMT, 1), blk, 0, stream>>>(
        aob, woT + (size_t)l * D * D, tmp, D, D, D, D, 0LL, 0LL, 0LL, 1.0f);
    add_res_k<<<2048, blk, 0, stream>>>(hbuf, tmp, (size_t)M * D);

    // FFN: h += (silu(fn@w1) * (fn@w3)) @ w2   (w1/w3 fused, z = 2)
    rmsnorm_k<<<M, blk, 0, stream>>>(hbuf, fnw + (size_t)l * D, fn);
    gemm_bt_k<<<dim3(FF / BNT, M / BMT, 2), blk, 0, stream>>>(
        fn, w13T + (size_t)l * 2 * FF * D, g1,
        D, D, D, FF, 0LL, (long long)FF * D, (long long)M * FF, 1.0f);
    silu_mul_k<<<4096, blk, 0, stream>>>(g1, g3, gs, (size_t)M * FF);
    gemm_bt_k<<<dim3(D / BNT, M / BMT, 1), blk, 0, stream>>>(
        gs, w2T + (size_t)l * D * FF, tmp, FF, FF, FF, D, 0LL, 0LL, 0LL, 1.0f);
    add_res_k<<<2048, blk, 0, stream>>>(hbuf, tmp, (size_t)M * D);
  }

  // final norm + logits = hn @ emb^T
  rmsnorm_k<<<M, blk, 0, stream>>>(hbuf, finw, hn);
  gemm_bt_k<<<dim3(V / BNT, M / BMT, 1), blk, 0, stream>>>(
      hn, embB, (float*)d_out, D, D, D, V, 0LL, 0LL, 0LL, 1.0f);
}
